// FlowmatchingActionHead_6597069767047
// MI455X (gfx1250) — matrix operand order repaired and checked by reading, not compiled, not run on hardware
//
#include <hip/hip_runtime.h>
#include <hip/hip_bf16.h>

// ---------------------------------------------------------------------------
// FlowmatchingActionHead on gfx1250 (MI455X), bf16 WMMA, f32 accumulate.
//   B=128, T=16, HIDDEN=1024, ACTION_DIM=32, NUM_EMB=32
//   one workgroup per batch element; 16 waves; each wave owns 64 output cols
// Memory-bound: ~388 MB distinct f32 weights @ 23.3 TB/s ~= 17 us; bf16 WMMA
// compute is a few us, so the goal is minimal per-load overhead:
//   - f32->bf16 pair pack = 2x v_add_u32 (+0x8000 round) + 1x v_perm_b32
//   - base-pointer + constant-IOFFSET addressing for all strided accesses
// ---------------------------------------------------------------------------

typedef __attribute__((ext_vector_type(16))) __bf16 v16bf;
typedef __attribute__((ext_vector_type(2)))  __bf16 v2bf;
typedef __attribute__((ext_vector_type(8)))  float  v8f;

union ABu { v16bf bf; unsigned u[8]; };

#define BATCH 128
#define TT    16
#define HD    1024
#define AD    32
#define SX    1026   // LDS row stride in ushorts (odd dword stride -> conflict-free A reads)

// Pack two f32 -> packed bf16x2 (lo in [15:0], hi in [31:16]).
// Round-to-nearest(half-up) via +0x8000, then one v_perm_b32 grabs both high
// halves: sel 0x07_06_03_02 -> {S0.b3, S0.b2, S1.b3, S1.b2} with S0=hi, S1=lo.
__device__ __forceinline__ unsigned pack2bf(float lo, float hi) {
#if __has_builtin(__builtin_amdgcn_cvt_pk_bf16_f32)
    v2bf p = __builtin_amdgcn_cvt_pk_bf16_f32(lo, hi);
    return __builtin_bit_cast(unsigned, p);
#else
    unsigned ul = __float_as_uint(lo) + 0x8000u;
    unsigned uh = __float_as_uint(hi) + 0x8000u;
    return __builtin_amdgcn_perm(uh, ul, 0x07060302u);
#endif
}

__device__ __forceinline__ unsigned short bf16_u16(float f) {
    return (unsigned short)((__float_as_uint(f) + 0x8000u) >> 16);
}

// A-fragment K offset for 16-bit 16x32 A tile (contiguous K order, matches B):
//   lanes 0-15 (g=0): VGPR v -> K=2v,2v+1
//   lanes 16-31 (g=1): K=16+2v,16+2v+1
__device__ __forceinline__ constexpr int a_koff(int v, int g) {
    return 16 * g + 2 * v;
}

// Load a 32(K) x 16(N) B-fragment from row-major f32 W (row stride HD floats),
// converting to bf16. Base pointer encodes (k0 + 16*g) row and column n, so all
// 16 loads get constant instruction offsets (max 15*HD*4 = 61440 B < 2^23).
__device__ __forceinline__ ABu load_b_frag(const float* __restrict__ wp) {
    ABu bb;
#pragma unroll
    for (int v = 0; v < 8; ++v)
        bb.u[v] = pack2bf(wp[(2 * v) * HD], wp[(2 * v + 1) * HD]);
    return bb;
}

__global__ __launch_bounds__(512)
void flow_head_kernel(const float* __restrict__ actions,   // (B,T,AD) f32
                      const int*   __restrict__ timesteps, // (B,)
                      const int*   __restrict__ cat_ids,   // (B,)
                      const float* __restrict__ W1,        // (E,AD,HD)
                      const float* __restrict__ b1,        // (E,HD)
                      const float* __restrict__ W2,        // (E,2HD,HD)
                      const float* __restrict__ b2,        // (E,HD)
                      const float* __restrict__ W3,        // (E,HD,HD)
                      const float* __restrict__ b3,        // (E,HD)
                      float*       __restrict__ out)       // (B,T,HD)
{
    __shared__ unsigned short xs[TT * SX];   // 32.8 KB: a_emb (bf16), later swish(h) (bf16)

    const int b    = blockIdx.x;
    const int tid  = threadIdx.x;
    const int lane = tid & 31;
    const int wave = tid >> 5;          // 0..15, owns columns [wave*64, wave*64+64)
    const int g    = lane >> 4;         // half-wave group
    const int lr   = lane & 15;         // A row  or  B/C/D column within tile
    const int nb   = wave * 64 + lr;    // column for j=0 tile

    const int   cat = cat_ids[b];
    const float tau = (float)timesteps[b];

    const float* W1e = W1 + (size_t)cat * AD * HD;
    const float* W2e = W2 + (size_t)cat * 2 * HD * HD;
    const float* W3e = W3 + (size_t)cat * HD * HD;
    const float* b1e = b1 + (size_t)cat * HD;
    const float* b2e = b2 + (size_t)cat * HD;
    const float* b3e = b3 + (size_t)cat * HD;

    // ---------------- Layer 1: a_emb = actions[b] (16x32) @ W1e (32x1024) + b1e
    ABu a1;
    {
        const float* arow = actions + ((size_t)b * TT + lr) * AD;
#pragma unroll
        for (int v = 0; v < 8; ++v) {
            const int k = a_koff(v, g);
            a1.u[v] = pack2bf(arow[k], arow[k + 1]);
        }
    }
    {
        const float*          w1p = W1e + (size_t)(16 * g) * HD + nb;
        unsigned short*       xsp = xs + 8 * g * SX + nb;
#pragma unroll
        for (int j = 0; j < 4; ++j) {
            const float bias = b1e[nb + j * 16];
            v8f acc;
#pragma unroll
            for (int r = 0; r < 8; ++r) acc[r] = bias;
            ABu bb = load_b_frag(w1p + j * 16);
            acc = __builtin_amdgcn_wmma_f32_16x16x32_bf16(false, a1.bf, false, bb.bf,
                                                          (short)0, acc, false, false);
#pragma unroll
            for (int r = 0; r < 8; ++r)            // C row m = r + 8*g, col n
                xsp[r * SX + j * 16] = bf16_u16(acc[r]);
        }
    }
    __syncthreads();

    // ---------------- Layer 2: h = [a_emb | pe(tau)] (16x2048) @ W2e + b2e ; swish
    v8f acc2[4];
#pragma unroll
    for (int j = 0; j < 4; ++j) {
        const float bias = b2e[nb + j * 16];
#pragma unroll
        for (int r = 0; r < 8; ++r) acc2[j][r] = bias;
    }
    const float neg_ln_per = -9.21034037197618f / 512.0f;   // -ln(10000)/half

    for (int k0 = 0; k0 < 2 * HD; k0 += 32) {
        ABu af;
        if (k0 < HD) {
            // a_emb half: packed bf16 pairs from LDS, constant ds offsets off one base
            const unsigned short* xrp = xs + lr * SX + k0;
#pragma unroll
            for (int v = 0; v < 8; ++v)
                af.u[v] = *(const unsigned*)&xrp[a_koff(v, g)];
        } else {
            // PE half: row-invariant, computed analytically (d even -> sin, odd -> cos)
            const int c0 = k0 - HD;
#pragma unroll
            for (int v = 0; v < 8; ++v) {
                const int c  = c0 + a_koff(v, g);     // always even
                const float arg = tau * __expf((float)(c >> 1) * neg_ln_per);
                af.u[v] = pack2bf(__sinf(arg), __cosf(arg));
            }
        }
        const float* w2p = W2e + (size_t)(k0 + 16 * g) * HD + nb;
#pragma unroll
        for (int j = 0; j < 4; ++j) {
            ABu bb = load_b_frag(w2p + j * 16);
            acc2[j] = __builtin_amdgcn_wmma_f32_16x16x32_bf16(false, af.bf, false, bb.bf,
                                                              (short)0, acc2[j], false, false);
        }
    }
    __syncthreads();    // everyone done reading a_emb from xs

    // swish(h) -> bf16 -> reuse xs
    {
        unsigned short* xsp = xs + 8 * g * SX + nb;
#pragma unroll
        for (int j = 0; j < 4; ++j) {
#pragma unroll
            for (int r = 0; r < 8; ++r) {
                const float h  = acc2[j][r];
                const float sw = h / (1.0f + __expf(-h));
                xsp[r * SX + j * 16] = bf16_u16(sw);
            }
        }
    }
    __syncthreads();

    // ---------------- Layer 3: out = swish(h) (16x1024) @ W3e + b3e
    v8f acc3[4];
#pragma unroll
    for (int j = 0; j < 4; ++j) {
        const float bias = b3e[nb + j * 16];
#pragma unroll
        for (int r = 0; r < 8; ++r) acc3[j][r] = bias;
    }
    for (int k0 = 0; k0 < HD; k0 += 32) {
        ABu af;
        const unsigned short* xrp = xs + lr * SX + k0;
#pragma unroll
        for (int v = 0; v < 8; ++v)
            af.u[v] = *(const unsigned*)&xrp[a_koff(v, g)];
        const float* w3p = W3e + (size_t)(k0 + 16 * g) * HD + nb;
#pragma unroll
        for (int j = 0; j < 4; ++j) {
            ABu bb = load_b_frag(w3p + j * 16);
            acc3[j] = __builtin_amdgcn_wmma_f32_16x16x32_bf16(false, af.bf, false, bb.bf,
                                                              (short)0, acc3[j], false, false);
        }
    }
    {
        float* op = out + ((size_t)b * TT + 8 * g) * HD + nb;
#pragma unroll
        for (int j = 0; j < 4; ++j)
#pragma unroll
            for (int r = 0; r < 8; ++r)
                op[(size_t)r * HD + j * 16] = acc3[j][r];
    }
}

extern "C" void kernel_launch(void* const* d_in, const int* in_sizes, int n_in,
                              void* d_out, int out_size, void* d_ws, size_t ws_size,
                              hipStream_t stream) {
    (void)in_sizes; (void)n_in; (void)out_size; (void)d_ws; (void)ws_size;
    const float* actions   = (const float*)d_in[0];
    const int*   timesteps = (const int*)  d_in[1];
    const int*   cat_ids   = (const int*)  d_in[2];
    const float* W1 = (const float*)d_in[3];
    const float* b1 = (const float*)d_in[4];
    const float* W2 = (const float*)d_in[5];
    const float* b2 = (const float*)d_in[6];
    const float* W3 = (const float*)d_in[7];
    const float* b3 = (const float*)d_in[8];
    float* out = (float*)d_out;

    dim3 grid(BATCH), block(512);
    hipLaunchKernelGGL(flow_head_kernel, grid, block, 0, stream,
                       actions, timesteps, cat_ids, W1, b1, W2, b2, W3, b3, out);
}
